// QuantLinear_128849019569
// MI455X (gfx1250) — matrix operand order repaired and checked by reading, not compiled, not run on hardware
//
#include <hip/hip_runtime.h>
#include <stdint.h>

// ---------------------------------------------------------------------------
// 4-bit group-quantized GEMM (GPTQ QuantLinear) for gfx1250 / MI455X.
// out[M,N] = x[M,K] @ dequant(qweight, qzeros, scales)[K,N] + bias
// M = 16384, K = N = 4096, group = 128, nibbles packed 8-per-int32.
//
// AI ~1000 FLOP/B -> compute-bound: ceiling is v_wmma_f32_16x16x32_bf16 rate.
// 256 threads (8 waves), 128x128 tile, K-step 64, double-buffered LDS in
// WMMA-fragment-major layout. All fragments of a stage are loaded into
// distinct registers BEFORE the next stage is staged, so the 16 WMMAs issue
// back-to-back behind a single partial s_wait_dscnt. bf16 pair packing is a
// single v_perm_b32; dequant is one v_fma per nibble (zero folded into -z*s).
// ---------------------------------------------------------------------------

typedef __attribute__((ext_vector_type(16))) __bf16 v16bf;
typedef __attribute__((ext_vector_type(8)))  float  v8f;

#define K_DIM   4096
#define N_DIM   4096
#define NPACK8  (N_DIM / 8)
#define BLK_M   128
#define BLK_N   128
#define BLK_K   64                       // two 32-K WMMA chunks per stage
#define KSTEPS  (K_DIM / BLK_K)          // 64

// fp32 pair -> packed bf16 dword via one v_perm_b32 (truncation; inputs carry
// 4-bit quant noise far above the 2^-9 relative truncation error)
__device__ __forceinline__ uint32_t bf16pack(float lo, float hi) {
    return __builtin_amdgcn_perm(__builtin_bit_cast(uint32_t, hi),
                                 __builtin_bit_cast(uint32_t, lo),
                                 0x07060302u);
}

__global__ __launch_bounds__(256)
void qlinear4_wmma_bf16(const float* __restrict__ x,
                        const int*   __restrict__ qweight,
                        const int*   __restrict__ qzeros,
                        const float* __restrict__ scales,
                        const float* __restrict__ bias,
                        float* __restrict__ out)
{
    // Fragment-major staging, chunk-major within a stage:
    // slot (c, t, lane) at dword offset ((c*256 + t*32 + lane) * 8) holds one
    // lane's 16-bf16 WMMA fragment (32 B) of K-chunk c.
    __shared__ __align__(16) uint32_t As[2][BLK_M * BLK_K / 2];  // 16 KB each
    __shared__ __align__(16) uint32_t Bs[2][BLK_K * BLK_N / 2];  // 16 KB each

    const int tid   = threadIdx.x;
    const int lane  = tid & 31;
    const int tile8 = tid >> 5;        // wave id == M-subtile; also loader tile
    const int half  = lane >> 4;       // lane half -> K sub-range of fragment
    const int l15   = lane & 15;

    const int bm = blockIdx.y * BLK_M;
    const int bn = blockIdx.x * BLK_N;

    // A fragment (ISA 7.12.2, 16-bit A 16x32): lanes 0-15 -> K {0..7,16..23},
    // lanes 16-31 -> K {8..15,24..31}; row M = l15.
    const float* xrow = x + (size_t)(bm + tile8 * 16 + l15) * K_DIM + (half ? 8 : 0);

    // B fragment (16-bit B 32x16): lanes 0-15 -> K {0..7,16..23}, lanes 16-31 -> K {8..15,24..31};
    // col N = l15.
    const int bcol = bn + tile8 * 16 + l15;
    const int bko  = half ? 8 : 0;

    const int slot = tile8 * 32 + lane;   // 0..255

    v8f acc[8];
    #pragma unroll
    for (int i = 0; i < 8; ++i)
        #pragma unroll
        for (int j = 0; j < 8; ++j) acc[i][j] = 0.0f;

    // ---- stage one 128x64 A tile (fp32 -> bf16), two fragment chunks ----
    auto loadA = [&](int kb, int s) {
        #pragma unroll
        for (int c = 0; c < 2; ++c) {
            const float4* p = reinterpret_cast<const float4*>(xrow + kb * BLK_K + c * 32);
            float4 c0 = p[0];            // K +0..3
            float4 c1 = p[1];            // K +4..7
            float4 c2 = p[4];            // K +16..19
            float4 c3 = p[5];            // K +20..23
            uint4* d = reinterpret_cast<uint4*>(&As[s][(c * 256 + slot) * 8]);
            d[0] = make_uint4(bf16pack(c0.x, c0.y), bf16pack(c0.z, c0.w),
                              bf16pack(c1.x, c1.y), bf16pack(c1.z, c1.w));
            d[1] = make_uint4(bf16pack(c2.x, c2.y), bf16pack(c2.z, c2.w),
                              bf16pack(c3.x, c3.y), bf16pack(c3.z, c3.w));
        }
    };

    // ---- stage one 64x128 dequantized B tile (one quant group per stage) ----
    auto loadB = [&](int kb, int s) {
        const int k0 = kb * BLK_K;
        const int g  = k0 >> 7;                       // k0 % 128 in {0,64} -> 1 group
        const float sc = scales[g * N_DIM + bcol];
        const uint32_t zw = (uint32_t)qzeros[g * NPACK8 + (bcol >> 3)];
        const float nzs = -(float)((zw >> ((bcol & 7) * 4)) & 15u) * sc;  // -z*s
        #pragma unroll
        for (int c = 0; c < 2; ++c) {
            const int r0 = (k0 + c * 32 + bko) >> 3;  // packed row: 8 K per word
            const uint32_t w0 = (uint32_t)qweight[(size_t)r0       * N_DIM + bcol];
            const uint32_t w1 = (uint32_t)qweight[(size_t)(r0 + 2) * N_DIM + bcol];
            uint32_t pk[8];
            #pragma unroll
            for (int p4 = 0; p4 < 4; ++p4) {
                // dequant = w * s + (-z*s): one v_bfe + v_cvt + v_fma per nibble
                float f0 = fmaf((float)((w0 >> (8 * p4))     & 15u), sc, nzs);
                float f1 = fmaf((float)((w0 >> (8 * p4 + 4)) & 15u), sc, nzs);
                pk[p4] = bf16pack(f0, f1);
                float f2 = fmaf((float)((w1 >> (8 * p4))     & 15u), sc, nzs);
                float f3 = fmaf((float)((w1 >> (8 * p4 + 4)) & 15u), sc, nzs);
                pk[4 + p4] = bf16pack(f2, f3);
            }
            uint4* d = reinterpret_cast<uint4*>(&Bs[s][(c * 256 + slot) * 8]);
            d[0] = make_uint4(pk[0], pk[1], pk[2], pk[3]);
            d[1] = make_uint4(pk[4], pk[5], pk[6], pk[7]);
        }
    };

    auto ldAfrag = [&](int s, int c) -> v16bf {
        return *reinterpret_cast<const v16bf*>(&As[s][(c * 256 + tile8 * 32 + lane) * 8]);
    };
    auto ldBfrag = [&](int s, int c, int nt) -> v16bf {
        return *reinterpret_cast<const v16bf*>(&Bs[s][(c * 256 + nt * 32 + lane) * 8]);
    };

    loadA(0, 0);
    loadB(0, 0);
    __syncthreads();

    #pragma unroll 1
    for (int kb = 0; kb < KSTEPS; ++kb) {
        const int cur = kb & 1;

        // 1) Issue ALL fragment loads of this stage into distinct registers.
        //    These ds_loads precede the staging ds_stores below, so the wait
        //    in front of the WMMA chain is partial (dscnt <= #stores), and the
        //    LDS latency hides under the global loads + dequant VALU.
        const v16bf a0 = ldAfrag(cur, 0);
        const v16bf a1 = ldAfrag(cur, 1);
        v16bf b0[8], b1[8];
        #pragma unroll
        for (int nt = 0; nt < 8; ++nt) {
            b0[nt] = ldBfrag(cur, 0, nt);
            b1[nt] = ldBfrag(cur, 1, nt);
        }

        // 2) Stage the next K-block into the other buffer.
        if (kb + 1 < KSTEPS) {
            if (kb + 2 < KSTEPS)
                __builtin_prefetch(xrow + (kb + 2) * BLK_K, 0, 0);  // global_prefetch
            loadA(kb + 1, cur ^ 1);
            loadB(kb + 1, cur ^ 1);
        }

        // 3) 16 back-to-back WMMAs; dependent writes to the same acc are
        //    8 WMMAs apart (chunk 0 sweep, then chunk 1 sweep).
        #pragma unroll
        for (int nt = 0; nt < 8; ++nt)
            acc[nt] = __builtin_amdgcn_wmma_f32_16x16x32_bf16(
                false, a0, false, b0[nt], (short)0, acc[nt], false, false);
        #pragma unroll
        for (int nt = 0; nt < 8; ++nt)
            acc[nt] = __builtin_amdgcn_wmma_f32_16x16x32_bf16(
                false, a1, false, b1[nt], (short)0, acc[nt], false, false);

        __syncthreads();   // next buffer visible; cur safe to overwrite next iter
    }

    // ---- epilogue: C layout (VGPR r -> row r / r+8 per lane half) + bias ----
    const int rbase = bm + tile8 * 16 + half * 8;
    #pragma unroll
    for (int nt = 0; nt < 8; ++nt) {
        const int col = bn + nt * 16 + l15;
        const float bv = bias[col];
        #pragma unroll
        for (int r = 0; r < 8; ++r)
            out[(size_t)(rbase + r) * N_DIM + col] = acc[nt][r] + bv;
    }
}

extern "C" void kernel_launch(void* const* d_in, const int* in_sizes, int n_in,
                              void* d_out, int out_size, void* d_ws, size_t ws_size,
                              hipStream_t stream) {
    const float* x       = (const float*)d_in[0];
    const int*   qweight = (const int*)  d_in[1];
    const int*   qzeros  = (const int*)  d_in[2];
    const float* scales  = (const float*)d_in[3];
    const float* bias    = (const float*)d_in[4];
    float*       out     = (float*)d_out;

    const int M = in_sizes[0] / K_DIM;            // 8*2048 = 16384
    dim3 grid(N_DIM / BLK_N, M / BLK_M);          // (32, 128)
    qlinear4_wmma_bf16<<<grid, dim3(256), 0, stream>>>(x, qweight, qzeros,
                                                       scales, bias, out);
}
